// PointNetPlusPlus_29978871726619
// MI455X (gfx1250) — matrix operand order repaired and checked by reading, not compiled, not run on hardware
//
#include <hip/hip_runtime.h>

typedef __attribute__((ext_vector_type(16))) _Float16 v16h;
typedef __attribute__((ext_vector_type(8)))  _Float16 v8h;
typedef __attribute__((ext_vector_type(8)))  float    v8f;

static constexpr int   Bc   = 32;
static constexpr int   Nc   = 4096;
static constexpr float EPSV = 1e-5f;
static constexpr float FINF = 3.0e38f;

// ---------------------------------------------------------------- split input
__global__ void k_split(const float* __restrict__ x, float* __restrict__ xyz,
                        float* __restrict__ pts, int total) {
  int t = blockIdx.x * blockDim.x + threadIdx.x;
  if (t >= total) return;
  const float* s = x + (size_t)t * 5;
  xyz[t * 3 + 0] = s[0]; xyz[t * 3 + 1] = s[1]; xyz[t * 3 + 2] = s[2];
  pts[t * 2 + 0] = s[3]; pts[t * 2 + 1] = s[4];
}

// -------------------------------------------------------- strided point sample
__global__ void k_sample(const float* __restrict__ src, float* __restrict__ dst,
                         int Nsrc, int S, int stride, int total) {
  int t = blockIdx.x * blockDim.x + threadIdx.x;
  if (t >= total) return;
  int b = t / S, s = t % S;
  int n = s * stride;
  const float* p = src + ((size_t)b * Nsrc + n) * 3;
  float* q = dst + (size_t)t * 3;
  q[0] = p[0]; q[1] = p[1]; q[2] = p[2];
}

// ------------------------------------------------- knn: one wave per query pt
__global__ void k_knn(const float* __restrict__ cloud, const float* __restrict__ query,
                      int* __restrict__ out, int Nsrc, int S, int K) {
  extern __shared__ float sd[];
  int q = blockIdx.x;          // b*S + s
  int b = q / S;
  int lane = threadIdx.x;
  const float* cb = cloud + (size_t)b * Nsrc * 3;
  float qx = query[(size_t)q * 3 + 0];
  float qy = query[(size_t)q * 3 + 1];
  float qz = query[(size_t)q * 3 + 2];
  for (int i = lane; i < Nsrc; i += 32) {
    float dx = cb[i * 3 + 0] - qx, dy = cb[i * 3 + 1] - qy, dz = cb[i * 3 + 2] - qz;
    sd[i] = dx * dx + dy * dy + dz * dz;
  }
  __syncthreads();
  for (int t = 0; t < K; ++t) {
    float bv = FINF; int bi = 0x7fffffff;
    for (int i = lane; i < Nsrc; i += 32) {
      float v = sd[i];
      if (v < bv || (v == bv && i < bi)) { bv = v; bi = i; }
    }
    for (int off = 16; off > 0; off >>= 1) {
      float ov = __shfl_down(bv, off);
      int   oi = __shfl_down(bi, off);
      if (ov < bv || (ov == bv && oi < bi)) { bv = ov; bi = oi; }
    }
    int best = __shfl(bi, 0);
    if (lane == 0) { out[(size_t)q * K + t] = best; sd[best] = FINF; }
    __syncthreads();
  }
}

// ------------------------------------- feature builders (write f16, pad zeros)
__global__ void k_feat_sa1(const float* __restrict__ xyz, const float* __restrict__ pts,
                           const float* __restrict__ nx, const int* __restrict__ knn,
                           _Float16* __restrict__ feat,
                           int Nsrc, int S, int NS, int CiPad, int total) {
  int t = blockIdx.x * blockDim.x + threadIdx.x;
  if (t >= total) return;
  int j = t % NS, s = (t / NS) % S, b = t / (NS * S);
  int p = (b * NS + j) * S + s;
  int idx = knn[((size_t)(b * S + s)) * NS + j];
  const float* cp = xyz + ((size_t)b * Nsrc + idx) * 3;
  const float* np = nx + ((size_t)(b * S + s)) * 3;
  _Float16* f = feat + (size_t)p * CiPad;
  f[0] = (_Float16)(cp[0] - np[0]);
  f[1] = (_Float16)(cp[1] - np[1]);
  f[2] = (_Float16)(cp[2] - np[2]);
  f[3] = (_Float16)pts[((size_t)b * Nsrc + idx) * 2 + 0];
  f[4] = (_Float16)pts[((size_t)b * Nsrc + idx) * 2 + 1];
  for (int c = 5; c < CiPad; ++c) f[c] = (_Float16)0.f;
}

__global__ void k_feat_sa2(const float* __restrict__ cloud, const float* __restrict__ p1,
                           const float* __restrict__ nx, const int* __restrict__ knn,
                           _Float16* __restrict__ feat,
                           int Nsrc, int S, int NS, int Cf, int CiPad, int total) {
  int t = blockIdx.x * blockDim.x + threadIdx.x;
  if (t >= total) return;
  int j = t % NS, s = (t / NS) % S, b = t / (NS * S);
  int p = (b * NS + j) * S + s;
  int idx = knn[((size_t)(b * S + s)) * NS + j];
  const float* cp = cloud + ((size_t)b * Nsrc + idx) * 3;
  const float* np = nx + ((size_t)(b * S + s)) * 3;
  _Float16* f = feat + (size_t)p * CiPad;
  f[0] = (_Float16)(cp[0] - np[0]);
  f[1] = (_Float16)(cp[1] - np[1]);
  f[2] = (_Float16)(cp[2] - np[2]);
  for (int c = 0; c < Cf; ++c)
    f[3 + c] = (_Float16)p1[((size_t)b * Cf + c) * Nsrc + idx];
  for (int c = 3 + Cf; c < CiPad; ++c) f[c] = (_Float16)0.f;
}

__global__ void k_feat_sa3(const float* __restrict__ p2, _Float16* __restrict__ feat,
                           int S, int Cf, int CiPad, int total) {
  int t = blockIdx.x * blockDim.x + threadIdx.x;  // b*S + s
  if (t >= total) return;
  int b = t / S, s = t % S;
  _Float16* f = feat + (size_t)t * CiPad;
  f[0] = (_Float16)0.f; f[1] = (_Float16)0.f; f[2] = (_Float16)0.f;  // gx == 0
  for (int c = 0; c < Cf; ++c)
    f[3 + c] = (_Float16)p2[((size_t)b * Cf + c) * S + s];
  for (int c = 3 + Cf; c < CiPad; ++c) f[c] = (_Float16)0.f;
}

// ------------------------------------------------------------- WMMA GEMM core
// W: [CoPad][CiPad] f16 row-major, X: [P][CiPad] f16 (position-major),
// Y: [CoPad][P] f32.
// One wave computes a 16(M) x 32(N) tile: 1 A-fragment reused by 2 WMMAs per
// K-step, 2-deep software pipeline (loads for step K+1 in flight during step
// K's WMMAs).  Live state ~80 VGPRs -> no scratch spills.
__global__ void __launch_bounds__(256)
k_gemm(const _Float16* __restrict__ W, const _Float16* __restrict__ X,
       float* __restrict__ Y, int CoPad, int CiPad, int P) {
  int wave = (blockIdx.x * blockDim.x + threadIdx.x) >> 5;
  int lane = threadIdx.x & 31;
  int nTM = CoPad >> 4, nTN = P >> 4;
  int nGN = (nTN + 1) >> 1;                      // groups of 2 N-tiles
  if (wave >= nTM * nGN) return;
  int tm = wave % nTM, gn = wave / nTM;
  int tn0 = gn * 2;
  int tcount = nTN - tn0; if (tcount > 2) tcount = 2;
  int r = lane & 15, hi = lane >> 4;

  const _Float16* wrow  = W + (size_t)(tm * 16 + r) * CiPad;
  const _Float16* xcol0 = X + (size_t)(tn0 * 16 + r) * CiPad;
  const _Float16* xcol1 = X + (size_t)((tn0 + (tcount > 1 ? 1 : 0)) * 16 + r) * CiPad;

  v8f acc0 = {}, acc1 = {};
  union F { v16h v; v8h h[2]; };
  F a0, a1, b00, b01, b10, b11;

  // A 16x32: lane(r,hi) holds K = 16hi..16hi+15
  // B 32x16: lane(col=r,hi) holds K = 16hi..16hi+15
#define LOAD_FRAG(A, B0, B1, K0)                                               \
  A.h[0]  = *(const v8h*)(wrow + (K0) + 16 * hi);                              \
  A.h[1]  = *(const v8h*)(wrow + (K0) + 16 * hi + 8);                          \
  B0.h[0] = *(const v8h*)(xcol0 + (K0) + 16 * hi);                             \
  B0.h[1] = *(const v8h*)(xcol0 + (K0) + 16 * hi + 8);                         \
  B1.h[0] = *(const v8h*)(xcol1 + (K0) + 16 * hi);                             \
  B1.h[1] = *(const v8h*)(xcol1 + (K0) + 16 * hi + 8);

#define WMMA2(A, B0, B1)                                                       \
  acc0 = __builtin_amdgcn_wmma_f32_16x16x32_f16(false, A.v, false, B0.v,       \
                                                (short)0, acc0, false, false); \
  acc1 = __builtin_amdgcn_wmma_f32_16x16x32_f16(false, A.v, false, B1.v,       \
                                                (short)0, acc1, false, false);

  LOAD_FRAG(a0, b00, b01, 0)
  int k0 = 0;
  for (;;) {
    int nk = k0 + 32;
    if (nk < CiPad) {
      LOAD_FRAG(a1, b10, b11, nk)
      __builtin_prefetch(xcol0 + nk + 32, 0, 3);
      WMMA2(a0, b00, b01)
    } else {
      WMMA2(a0, b00, b01)
      break;
    }
    k0 = nk; nk = k0 + 32;
    if (nk < CiPad) {
      LOAD_FRAG(a0, b00, b01, nk)
      __builtin_prefetch(xcol0 + nk + 32, 0, 3);
      WMMA2(a1, b10, b11)
    } else {
      WMMA2(a1, b10, b11)
      break;
    }
    k0 = nk;
  }
#undef LOAD_FRAG
#undef WMMA2

  // D: VGPR v, lane(r,hi) -> row = 16*tm + v + 8*hi, col = 16*tn + r
  float* y0 = Y + (size_t)(tm * 16 + 8 * hi) * P + tn0 * 16 + r;
#pragma unroll
  for (int v = 0; v < 8; ++v) y0[(size_t)v * P] = acc0[v];
  if (tcount > 1) {
    float* y1 = y0 + 16;
#pragma unroll
    for (int v = 0; v < 8; ++v) y1[(size_t)v * P] = acc1[v];
  }
}

// ------------------------------------------------------------------ batch norm
__global__ void k_bn_stats(const float* __restrict__ Y, float* __restrict__ mean,
                           float* __restrict__ rsig, int P) {
  __shared__ float ss[256], sq[256];
  int c = blockIdx.x;
  const float* row = Y + (size_t)c * P;
  float s = 0.f, q = 0.f;
  for (int i = threadIdx.x; i < P; i += blockDim.x) { float v = row[i]; s += v; q += v * v; }
  ss[threadIdx.x] = s; sq[threadIdx.x] = q;
  __syncthreads();
  for (int off = 128; off > 0; off >>= 1) {
    if ((int)threadIdx.x < off) { ss[threadIdx.x] += ss[threadIdx.x + off];
                                  sq[threadIdx.x] += sq[threadIdx.x + off]; }
    __syncthreads();
  }
  if (threadIdx.x == 0) {
    float m = ss[0] / P;
    float var = sq[0] / P - m * m;
    mean[c] = m; rsig[c] = rsqrtf(var + EPSV);
  }
}

// BN apply + optional ReLU, restage to f16 position-major for next GEMM
__global__ void k_bn_apply_h(const float* __restrict__ Y, const float* __restrict__ mean,
                             const float* __restrict__ rsig, const float* __restrict__ g,
                             const float* __restrict__ be, _Float16* __restrict__ Xn,
                             int C, int P, int relu) {
  int t = blockIdx.x * blockDim.x + threadIdx.x;
  if (t >= C * P) return;
  int c = t / P, p = t % P;
  float v = (Y[t] - mean[c]) * rsig[c] * g[c] + be[c];
  if (relu) v = fmaxf(v, 0.f);
  Xn[(size_t)p * C + c] = (_Float16)v;
}

// BN apply in-place f32 (last conv of a stage, no ReLU)
__global__ void k_bn_apply_f(float* __restrict__ Y, const float* __restrict__ mean,
                             const float* __restrict__ rsig, const float* __restrict__ g,
                             const float* __restrict__ be, int C, int P) {
  int t = blockIdx.x * blockDim.x + threadIdx.x;
  if (t >= C * P) return;
  int c = t / P;
  Y[t] = (Y[t] - mean[c]) * rsig[c] * g[c] + be[c];
}

// max over nsample -> pooled[b][c][s]
__global__ void k_maxpool(const float* __restrict__ Y, float* __restrict__ pooled,
                          int C, int NS, int S, int P, int total) {
  int t = blockIdx.x * blockDim.x + threadIdx.x;
  if (t >= total) return;
  int s = t % S, c = (t / S) % C, b = t / (S * C);
  float m = -FINF;
  for (int j = 0; j < NS; ++j)
    m = fmaxf(m, Y[(size_t)c * P + (size_t)(b * NS + j) * S + s]);
  pooled[((size_t)b * C + c) * S + s] = m;
}

// SA3 pool over points: Y[1024][B*128] -> Xfc[b][c] f16
__global__ void k_pool3(const float* __restrict__ Y, _Float16* __restrict__ Xfc,
                        int C, int S, int total) {
  int t = blockIdx.x * blockDim.x + threadIdx.x;   // b*C + c
  if (t >= total) return;
  int b = t / C, c = t % C;
  float m = -FINF;
  for (int s = 0; s < S; ++s)
    m = fmaxf(m, Y[(size_t)c * (Bc * S) + b * S + s]);
  Xfc[t] = (_Float16)m;
}

// ------------------------------------------------------- weight f32->f16 (pad)
__global__ void k_wcvt(const float* __restrict__ src, _Float16* __restrict__ dst,
                       int Co, int Ci, int CoPad, int CiPad) {
  int t = blockIdx.x * blockDim.x + threadIdx.x;
  if (t >= CoPad * CiPad) return;
  int r = t / CiPad, c = t % CiPad;
  dst[t] = (r < Co && c < Ci) ? (_Float16)src[(size_t)r * Ci + c] : (_Float16)0.f;
}

// ------------------------------------------------------------------ final head
__global__ void k_final(const float* __restrict__ Y, const float* __restrict__ b3,
                        float* __restrict__ out) {
  int t = blockIdx.x * blockDim.x + threadIdx.x;
  if (t >= Bc * 12) return;
  int b = t / 12, o = t % 12;
  out[t] = Y[(size_t)o * Bc + b] + b3[o];
}

// =============================================================================
extern "C" void kernel_launch(void* const* d_in, const int* in_sizes, int n_in,
                              void* d_out, int out_size, void* d_ws, size_t ws_size,
                              hipStream_t stream) {
  (void)in_sizes; (void)n_in; (void)out_size; (void)ws_size;
  const float* x = (const float*)d_in[0];
  struct CB { const float *w, *b, *g, *be; };
  CB L[11];
  int pi = 1;
  for (int i = 0; i < 11; ++i) {
    L[i].w  = (const float*)d_in[pi++];
    L[i].b  = (const float*)d_in[pi++];
    L[i].g  = (const float*)d_in[pi++];
    L[i].be = (const float*)d_in[pi++];
  }
  const float* fc3w = (const float*)d_in[pi++];
  const float* fc3b = (const float*)d_in[pi++];

  // ---- layer geometry: {Co, Ci, CoPad, CiPad}
  static const int Co[12]    = { 64, 64, 128, 128, 128, 256, 256, 512, 1024, 512, 256, 12 };
  static const int Ci[12]    = {  5, 64,  64, 131, 128, 128, 259, 256,  512, 1024, 512, 256 };
  static const int CoPad[12] = { 64, 64, 128, 128, 128, 256, 256, 512, 1024, 512, 256, 16 };
  static const int CiPad[12] = { 32, 64,  64, 160, 128, 128, 288, 256,  512, 1024, 512, 256 };

  const int S1 = 512, NS1 = 32, S2 = 128, NS2 = 64, S3 = 128;
  const int P1 = Bc * NS1 * S1;   // 524288
  const int P2 = Bc * NS2 * S2;   // 262144
  const int P3 = Bc * S3;         // 4096

  // ---- workspace carve
  char* wsb = (char*)d_ws; size_t off = 0;
  auto alloc = [&](size_t bytes) -> void* {
    void* p = wsb + off; off = (off + bytes + 255) & ~(size_t)255; return p;
  };
  float* xyz  = (float*)alloc((size_t)Bc * Nc * 3 * 4);
  float* pts  = (float*)alloc((size_t)Bc * Nc * 2 * 4);
  float* nx1  = (float*)alloc((size_t)Bc * S1 * 3 * 4);
  float* nx2  = (float*)alloc((size_t)Bc * S2 * 3 * 4);
  int*   idx1 = (int*)  alloc((size_t)Bc * S1 * NS1 * 4);
  int*   idx2 = (int*)  alloc((size_t)Bc * S2 * NS2 * 4);
  float* p1   = (float*)alloc((size_t)Bc * 128 * S1 * 4);
  float* p2   = (float*)alloc((size_t)Bc * 256 * S2 * 4);
  float* mean = (float*)alloc(1024 * 4);
  float* rsig = (float*)alloc(1024 * 4);
  _Float16* Wf[12];
  for (int i = 0; i < 12; ++i)
    Wf[i] = (_Float16*)alloc((size_t)CoPad[i] * CiPad[i] * 2);
  _Float16* Xb = (_Float16*)alloc((size_t)160 * P2 * 2);   // max X staging (84MB)
  float*    Yb = (float*)   alloc((size_t)128 * P1 * 4);   // max Y (268MB)

  auto cdiv = [](int a, int b) { return (a + b - 1) / b; };

  // ---- convert all weights (fc3 is layer 11)
  for (int i = 0; i < 11; ++i)
    k_wcvt<<<cdiv(CoPad[i] * CiPad[i], 256), 256, 0, stream>>>(
        L[i].w, Wf[i], Co[i], Ci[i], CoPad[i], CiPad[i]);
  k_wcvt<<<cdiv(CoPad[11] * CiPad[11], 256), 256, 0, stream>>>(
      fc3w, Wf[11], Co[11], Ci[11], CoPad[11], CiPad[11]);

  auto gemm = [&](int li, const _Float16* X, int P) {
    int nTM = CoPad[li] / 16, nTN = P / 16;
    int waves = nTM * cdiv(nTN, 2);
    k_gemm<<<cdiv(waves, 8), 256, 0, stream>>>(Wf[li], X, Yb, CoPad[li], CiPad[li], P);
  };
  auto bn_h = [&](int li, int P, int relu) {    // stats + apply -> f16 restage
    k_bn_stats<<<Co[li], 256, 0, stream>>>(Yb, mean, rsig, P);
    k_bn_apply_h<<<cdiv(Co[li] * P, 256), 256, 0, stream>>>(
        Yb, mean, rsig, L[li].g, L[li].be, Xb, Co[li], P, relu);
  };
  auto bn_f = [&](int li, int P) {              // stats + apply in-place f32
    k_bn_stats<<<Co[li], 256, 0, stream>>>(Yb, mean, rsig, P);
    k_bn_apply_f<<<cdiv(Co[li] * P, 256), 256, 0, stream>>>(
        Yb, mean, rsig, L[li].g, L[li].be, Co[li], P);
  };

  // ---- stage 0: split input
  k_split<<<cdiv(Bc * Nc, 256), 256, 0, stream>>>(x, xyz, pts, Bc * Nc);

  // ---- SA1
  k_sample<<<cdiv(Bc * S1, 256), 256, 0, stream>>>(xyz, nx1, Nc, S1, Nc / S1, Bc * S1);
  k_knn<<<Bc * S1, 32, Nc * sizeof(float), stream>>>(xyz, nx1, idx1, Nc, S1, NS1);
  k_feat_sa1<<<cdiv(P1, 256), 256, 0, stream>>>(xyz, pts, nx1, idx1, Xb, Nc, S1, NS1, 32, P1);
  gemm(0, Xb, P1); bn_h(0, P1, 1);
  gemm(1, Xb, P1); bn_h(1, P1, 1);
  gemm(2, Xb, P1); bn_f(2, P1);
  k_maxpool<<<cdiv(Bc * 128 * S1, 256), 256, 0, stream>>>(Yb, p1, 128, NS1, S1, P1, Bc * 128 * S1);

  // ---- SA2
  k_sample<<<cdiv(Bc * S2, 256), 256, 0, stream>>>(nx1, nx2, S1, S2, S1 / S2, Bc * S2);
  k_knn<<<Bc * S2, 32, S1 * sizeof(float), stream>>>(nx1, nx2, idx2, S1, S2, NS2);
  k_feat_sa2<<<cdiv(P2, 256), 256, 0, stream>>>(nx1, p1, nx2, idx2, Xb, S1, S2, NS2, 128, 160, P2);
  gemm(3, Xb, P2); bn_h(3, P2, 1);
  gemm(4, Xb, P2); bn_h(4, P2, 1);
  gemm(5, Xb, P2); bn_f(5, P2);
  k_maxpool<<<cdiv(Bc * 256 * S2, 256), 256, 0, stream>>>(Yb, p2, 256, NS2, S2, P2, Bc * 256 * S2);

  // ---- SA3 (nsample=1: nearest neighbor is self -> gx==0, gp==p2)
  k_feat_sa3<<<cdiv(P3, 256), 256, 0, stream>>>(p2, Xb, S3, 256, 288, P3);
  gemm(6, Xb, P3); bn_h(6, P3, 1);
  gemm(7, Xb, P3); bn_h(7, P3, 1);
  gemm(8, Xb, P3); bn_f(8, P3);
  k_pool3<<<cdiv(Bc * 1024, 256), 256, 0, stream>>>(Yb, Xb, 1024, S3, Bc * 1024);

  // ---- FC head (P = B = 32)
  gemm(9,  Xb, Bc); bn_h(9,  Bc, 1);
  gemm(10, Xb, Bc); bn_h(10, Bc, 1);
  gemm(11, Xb, Bc);
  k_final<<<cdiv(Bc * 12, 256), 256, 0, stream>>>(Yb, fc3b, (float*)d_out);
}
